// Visibility_Head_28166395527476
// MI455X (gfx1250) — matrix operand order repaired and checked by reading, not compiled, not run on hardware
//
#include <hip/hip_runtime.h>
#include <hip/hip_bf16.h>

// ---------------------------------------------------------------------------
// Visibility head (deformable attention) for MI455X / gfx1250, wave32 + WMMA.
//   prep       : convert q_t->bf16, transpose+convert W_value/[W_off|W_attn]/W_out
//   pyr01      : level0 copy + level1 2x2-avg fused (one read of h_t)
//   pyr23      : levels 2+3 from f32 level-1 buffer
//   gemm_n256  : value = f @ Wv + b   (M=130560,K=256,N=256) -> bf16
//   gemm_n384  : [off|attn] = q @ [Woff|Wattn] + b (M=6400,N=384)
//   locaw      : softmax over 16 pts/head + sampling locations
//   sample     : bilinear gather-weighted sum (wave per (b,n,head), lane=chan)
//   gemm_n256  : attn = samp @ Wout + b (M=6400,N=256) -> f32
//   finalize   : residual + LayerNorm + vis/unc projections (wave per row)
// ---------------------------------------------------------------------------

typedef __bf16 bf16;
typedef __attribute__((ext_vector_type(16))) __bf16 v16bf;
typedef __attribute__((ext_vector_type(8)))  __bf16 v8bf;
typedef __attribute__((ext_vector_type(8)))  float  v8f;

#define BB     8
#define NN     800
#define CC     256
#define BN     (BB * NN)          // 6400
#define S_TOT  16320
#define HP     96
#define WP     128
#define MROWS  (BB * S_TOT)       // 130560  (value GEMM rows)

// ---------------- WMMA fragment loaders (bf16, K-stride 256) ----------------

__device__ __forceinline__ v16bf load_a_frag(const bf16* A, int m0, int kb, int lane) {
  // A: row-major [M,256] bf16. 16-bit A 16x32 layout (ISA 7.12.2):
  // lane<16 -> elems {K..K+7, K+16..K+23}; lane>=16 -> +8.
  const int row  = m0 + (lane & 15);
  const int koff = kb + ((lane >> 4) << 3);
  const bf16* p = A + (size_t)row * 256 + koff;
  v8bf lo = *(const v8bf*)(p);
  v8bf hi = *(const v8bf*)(p + 16);
  return __builtin_shufflevector(lo, hi, 0,1,2,3,4,5,6,7,8,9,10,11,12,13,14,15);
}

__device__ __forceinline__ v16bf load_b_frag(const bf16* Wt, int n0, int kb, int lane) {
  // Wt: row-major [N,256] bf16 (original W transposed). B 32x16 layout:
  // lane -> col, elems -> K (lane<16: kb+0..7, kb+16..23; lane>=16: +8).
  const int col = n0 + (lane & 15);
  const int kk  = kb + ((lane >> 4) << 3);
  const bf16* p = Wt + (size_t)col * 256 + kk;
  v8bf lo = *(const v8bf*)(p);
  v8bf hi = *(const v8bf*)(p + 16);
  return __builtin_shufflevector(lo, hi, 0,1,2,3,4,5,6,7,8,9,10,11,12,13,14,15);
}

__device__ __forceinline__ v8f wmma_bf16(v16bf a, v16bf b, v8f c) {
  return __builtin_amdgcn_wmma_f32_16x16x32_bf16(false, a, false, b,
                                                 (short)0, c, false, false);
}

// ------------------------- GEMM: N=256, K=256 -------------------------------
// Block = 128 threads = 4 waves. Each wave computes a 32M x 64N tile
// (2 A-fragments x 4 B-fragments -> 8 accumulators), so every B fragment
// feeds two WMMAs. Block covers 32M x 256N; grid.x = M/32.
template <typename OutT>
__global__ __launch_bounds__(128) void gemm_n256(const bf16* __restrict__ A,
                                                 const bf16* __restrict__ Wt,
                                                 const float* __restrict__ bias,
                                                 OutT* __restrict__ D) {
  const int lane = threadIdx.x & 31;
  const int wave = threadIdx.x >> 5;
  const int m0   = blockIdx.x * 32;
  const int n0   = wave * 64;

  // Warm L2 for the next block's A rows (global_prefetch_b8; speculative).
  __builtin_prefetch(A + (size_t)(m0 + 32 + (lane & 15)) * 256, 0, 1);

  v8f acc0[4] = {};
  v8f acc1[4] = {};
#pragma unroll
  for (int kb = 0; kb < 256; kb += 32) {
    v16bf a0 = load_a_frag(A, m0,      kb, lane);
    v16bf a1 = load_a_frag(A, m0 + 16, kb, lane);
#pragma unroll
    for (int t = 0; t < 4; ++t) {
      v16bf b = load_b_frag(Wt, n0 + t * 16, kb, lane);
      acc0[t] = wmma_bf16(a0, b, acc0[t]);
      acc1[t] = wmma_bf16(a1, b, acc1[t]);
    }
  }
  const int rsub = (lane >> 4) << 3;          // row sub-offset 0 or 8
#pragma unroll
  for (int t = 0; t < 4; ++t) {
    const int c = n0 + t * 16 + (lane & 15);
    const float bv = bias[c];
    OutT* dp0 = D + (size_t)(m0 + rsub) * 256 + c;
    OutT* dp1 = D + (size_t)(m0 + 16 + rsub) * 256 + c;
#pragma unroll
    for (int j = 0; j < 8; ++j) {
      dp0[(size_t)j * 256] = (OutT)(acc0[t][j] + bv);
      dp1[(size_t)j * 256] = (OutT)(acc1[t][j] + bv);
    }
  }
}

// ------------------------- GEMM: N=384, K=256 -------------------------------
// Block = 192 threads = 6 waves, one m-tile per block, wave w covers cols 64w.
__global__ __launch_bounds__(192) void gemm_n384(const bf16* __restrict__ A,
                                                 const bf16* __restrict__ Wt,
                                                 const float* __restrict__ bias0, // 256
                                                 const float* __restrict__ bias1, // 128
                                                 float* __restrict__ D) {
  const int lane = threadIdx.x & 31;
  const int wave = threadIdx.x >> 5;
  const int m0   = blockIdx.x * 16;
  const int n0   = wave * 64;

  v8f acc[4] = {};
#pragma unroll
  for (int kb = 0; kb < 256; kb += 32) {
    v16bf a = load_a_frag(A, m0, kb, lane);
#pragma unroll
    for (int t = 0; t < 4; ++t) {
      v16bf b = load_b_frag(Wt, n0 + t * 16, kb, lane);
      acc[t] = wmma_bf16(a, b, acc[t]);
    }
  }
  const int rsub = (lane >> 4) << 3;
#pragma unroll
  for (int t = 0; t < 4; ++t) {
    const int c = n0 + t * 16 + (lane & 15);
    const float bv = (c < 256) ? bias0[c] : bias1[c - 256];
    float* dp = D + (size_t)(m0 + rsub) * 384 + c;
#pragma unroll
    for (int j = 0; j < 8; ++j) dp[(size_t)j * 384] = acc[t][j] + bv;
  }
}

// ------------------------- prep: converts + transposes ----------------------
__global__ void prep(const float* __restrict__ q,    // [6400,256]
                     const float* __restrict__ Wv,   // [256,256]
                     const float* __restrict__ Woff, // [256,256]
                     const float* __restrict__ Wattn,// [256,128]
                     const float* __restrict__ Wout, // [256,256]
                     bf16* __restrict__ qb, bf16* __restrict__ Wvt,
                     bf16* __restrict__ Woat, bf16* __restrict__ Woutt) {
  const int QN = BN * CC;                 // 1,638,400
  const int total = QN + 65536 + 98304 + 65536;
  for (int i = blockIdx.x * blockDim.x + threadIdx.x; i < total;
       i += gridDim.x * blockDim.x) {
    if (i < QN) {
      qb[i] = (bf16)q[i];
    } else if (i < QN + 65536) {
      int j = i - QN, n = j >> 8, k = j & 255;
      Wvt[j] = (bf16)Wv[k * 256 + n];
    } else if (i < QN + 65536 + 98304) {
      int j = i - QN - 65536, n = j >> 8, k = j & 255;
      Woat[j] = (bf16)((n < 256) ? Woff[k * 256 + n] : Wattn[k * 128 + (n - 256)]);
    } else {
      int j = i - QN - 65536 - 98304, n = j >> 8, k = j & 255;
      Woutt[j] = (bf16)Wout[k * 256 + n];
    }
  }
}

// ---------------- pyramid levels 0+1 (single pass over h_t) -----------------
// Thread per (b, y1, x1, c): reads the 2x2 level-0 block once, emits 4 bf16
// level-0 elements, 1 bf16 level-1 element, and f32 level-1 for levels 2/3.
__global__ void pyr01(const float* __restrict__ h, bf16* __restrict__ f,
                      float* __restrict__ l1) {
  const int idx = blockIdx.x * blockDim.x + threadIdx.x;   // exact: 8*3072*256
  const int b = idx / (3072 * 256);
  const int r = idx % (3072 * 256);
  const int s1 = r >> 8, c = r & 255;
  const int y1 = s1 >> 6, x1 = s1 & 63;                    // level-1 grid 48x64
  const size_t base = ((size_t)b * (HP * WP) + (2 * y1) * WP + 2 * x1) * 256 + c;
  const float v00 = h[base];
  const float v01 = h[base + 256];
  const float v10 = h[base + (size_t)WP * 256];
  const float v11 = h[base + (size_t)WP * 256 + 256];
  const size_t f0 = ((size_t)b * S_TOT + (2 * y1) * WP + 2 * x1) * 256 + c;
  f[f0]                       = (bf16)v00;
  f[f0 + 256]                 = (bf16)v01;
  f[f0 + (size_t)WP * 256]       = (bf16)v10;
  f[f0 + (size_t)WP * 256 + 256] = (bf16)v11;
  const float m = (v00 + v01 + v10 + v11) * 0.25f;
  f[((size_t)b * S_TOT + 12288 + s1) * 256 + c] = (bf16)m;
  l1[((size_t)b * 3072 + s1) * 256 + c] = m;
}

// ---------------- pyramid levels 2+3 (from f32 level-1) ---------------------
// Thread per (b, y3, x3, c): covers a 2x2 level-2 patch (4x4 level-1 reads).
__global__ void pyr23(const float* __restrict__ l1, bf16* __restrict__ f) {
  const int idx = blockIdx.x * blockDim.x + threadIdx.x;   // exact: 8*192*256
  const int b = idx / (192 * 256);
  const int r = idx % (192 * 256);
  const int s3 = r >> 8, c = r & 255;
  const int y3 = s3 >> 4, x3 = s3 & 15;                    // level-3 grid 12x16
  float m3 = 0.f;
#pragma unroll
  for (int dy = 0; dy < 2; ++dy) {
#pragma unroll
    for (int dx = 0; dx < 2; ++dx) {
      const int y2 = 2 * y3 + dy, x2 = 2 * x3 + dx;        // level-2 grid 24x32
      float m2 = 0.f;
#pragma unroll
      for (int i = 0; i < 2; ++i)
#pragma unroll
        for (int j = 0; j < 2; ++j)
          m2 += l1[((size_t)b * 3072 + (2 * y2 + i) * 64 + (2 * x2 + j)) * 256 + c];
      m2 *= 0.25f;
      f[((size_t)b * S_TOT + 15360 + y2 * 32 + x2) * 256 + c] = (bf16)m2;
      m3 += m2;
    }
  }
  f[((size_t)b * S_TOT + 16128 + s3) * 256 + c] = (bf16)(m3 * 0.25f);
}

// ---------------- softmax over 16 pts + sampling locations ------------------
__global__ void locaw(const float* __restrict__ oa,  // [6400,384]
                      const float* __restrict__ tc,  // [6400,2]
                      float* __restrict__ aw,        // [6400,128]
                      float* __restrict__ loc) {     // [6400,8,16,2]
  const int t = blockIdx.x * blockDim.x + threadIdx.x;      // 51200 exact
  const int r = t >> 3, hh = t & 7;
  const float lw[4] = {128.f, 64.f, 32.f, 16.f};
  const float lh[4] = {96.f, 48.f, 24.f, 12.f};
  float rx = tc[r * 2 + 0] * (1.f / 512.f);
  float ry = tc[r * 2 + 1] * (1.f / 384.f);
  rx = fminf(fmaxf(rx, 0.f), 1.f);
  ry = fminf(fmaxf(ry, 0.f), 1.f);
  const float* lg = oa + (size_t)r * 384 + 256 + hh * 16;
  float m = -3.4e38f;
#pragma unroll
  for (int j = 0; j < 16; ++j) m = fmaxf(m, lg[j]);
  float e[16], sum = 0.f;
#pragma unroll
  for (int j = 0; j < 16; ++j) { e[j] = __expf(lg[j] - m); sum += e[j]; }
  const float inv = 1.f / sum;
  const float* ob = oa + (size_t)r * 384 + hh * 32;
  float* lp = loc + (size_t)(r * 8 + hh) * 32;
#pragma unroll
  for (int j = 0; j < 16; ++j) {
    aw[(size_t)r * 128 + hh * 16 + j] = e[j] * inv;
    const int l = j >> 2;
    lp[2 * j + 0] = rx + ob[2 * j + 0] / lw[l];
    lp[2 * j + 1] = ry + ob[2 * j + 1] / lh[l];
  }
}

// ---------------------- bilinear deformable sampling ------------------------
// One wave per (b,n,head); lane = channel within head (DH=32).
__global__ __launch_bounds__(256) void sample(const bf16* __restrict__ value, // [B*S,256]
                                              const float* __restrict__ loc,
                                              const float* __restrict__ aw,
                                              bf16* __restrict__ samp) {      // [6400,256]
  const int lane = threadIdx.x & 31;
  const int wid  = blockIdx.x * 8 + (threadIdx.x >> 5);     // 0..51199
  const int hh = wid & 7, r = wid >> 3, b = r / NN;
  const int lw[4] = {128, 64, 32, 16};
  const int lhh[4] = {96, 48, 24, 12};
  const int loff[4] = {0, 12288, 15360, 16128};
  const float* lp = loc + (size_t)(r * 8 + hh) * 32;
  const float* ap = aw + (size_t)r * 128 + hh * 16;
  float acc = 0.f;
#pragma unroll
  for (int j = 0; j < 16; ++j) {
    const int l = j >> 2;
    const int W = lw[l], H = lhh[l];
    const float x = lp[2 * j + 0] * (float)W - 0.5f;
    const float y = lp[2 * j + 1] * (float)H - 0.5f;
    const float awj = ap[j];
    const float x0f = floorf(x), y0f = floorf(y);
    const float fx = x - x0f, fy = y - y0f;
    const int x0 = (int)x0f, y0 = (int)y0f;
#pragma unroll
    for (int dy = 0; dy < 2; ++dy) {
#pragma unroll
      for (int dx = 0; dx < 2; ++dx) {
        const int xi = x0 + dx, yi = y0 + dy;
        const float wgt = (dx ? fx : 1.f - fx) * (dy ? fy : 1.f - fy);
        const bool valid = (xi >= 0) & (xi < W) & (yi >= 0) & (yi < H);
        const int ix = min(max(xi, 0), W - 1);
        const int iy = min(max(yi, 0), H - 1);
        const float wv = valid ? wgt * awj : 0.f;
        const size_t vidx = ((size_t)(b * S_TOT + loff[l] + iy * W + ix)) * 256 + hh * 32 + lane;
        acc += wv * (float)value[vidx];
      }
    }
  }
  samp[(size_t)r * 256 + hh * 32 + lane] = (bf16)acc;
}

// -------------- residual + LayerNorm + vis/unc projections ------------------
__global__ __launch_bounds__(256) void finalize(const float* __restrict__ q,
                                                const float* __restrict__ attn,
                                                const float* __restrict__ lng,
                                                const float* __restrict__ lnb,
                                                const float* __restrict__ Wvis,
                                                const float* __restrict__ bvis,
                                                const float* __restrict__ Wunc,
                                                const float* __restrict__ bunc,
                                                float* __restrict__ out) {
  const int lane = threadIdx.x & 31;
  const int r = blockIdx.x * 8 + (threadIdx.x >> 5);        // 0..6399
  const float* qp = q + (size_t)r * 256;
  const float* ap = attn + (size_t)r * 256;
  float qv[8], hv[8], s = 0.f;
#pragma unroll
  for (int i = 0; i < 8; ++i) {
    const int c = lane + 32 * i;
    qv[i] = qp[c];
    hv[i] = qv[i] + ap[c];
    s += hv[i];
  }
#pragma unroll
  for (int m = 16; m >= 1; m >>= 1) s += __shfl_xor(s, m, 32);
  const float mu = s * (1.f / 256.f);
  float v = 0.f;
#pragma unroll
  for (int i = 0; i < 8; ++i) { const float d = hv[i] - mu; v += d * d; }
#pragma unroll
  for (int m = 16; m >= 1; m >>= 1) v += __shfl_xor(v, m, 32);
  const float inv = rsqrtf(v * (1.f / 256.f) + 1e-5f);
  float vl = 0.f, ul = 0.f;
#pragma unroll
  for (int i = 0; i < 8; ++i) {
    const int c = lane + 32 * i;
    const float ql = lng[c] * (hv[i] - mu) * inv + lnb[c];
    vl += qv[i] * Wvis[c] + ql * Wvis[256 + c];
    ul += qv[i] * Wunc[c] + ql * Wunc[256 + c];
  }
#pragma unroll
  for (int m = 16; m >= 1; m >>= 1) {
    vl += __shfl_xor(vl, m, 32);
    ul += __shfl_xor(ul, m, 32);
  }
  if (lane == 0) {
    out[r] = vl + bvis[0];
    out[BN + r] = ul + bunc[0];
  }
}

// ---------------------------------------------------------------------------
extern "C" void kernel_launch(void* const* d_in, const int* in_sizes, int n_in,
                              void* d_out, int out_size, void* d_ws, size_t ws_size,
                              hipStream_t stream) {
  const float* q_t   = (const float*)d_in[0];
  const float* h_t   = (const float*)d_in[1];
  const float* tc    = (const float*)d_in[2];
  const float* Wv    = (const float*)d_in[3];
  const float* bv    = (const float*)d_in[4];
  const float* Woff  = (const float*)d_in[5];
  const float* boff  = (const float*)d_in[6];
  const float* Wattn = (const float*)d_in[7];
  const float* battn = (const float*)d_in[8];
  const float* Wout  = (const float*)d_in[9];
  const float* bout  = (const float*)d_in[10];
  const float* lng   = (const float*)d_in[11];
  const float* lnb   = (const float*)d_in[12];
  const float* Wvis  = (const float*)d_in[13];
  const float* bvis  = (const float*)d_in[14];
  const float* Wunc  = (const float*)d_in[15];
  const float* bunc  = (const float*)d_in[16];
  float* out = (float*)d_out;

  // ---- workspace layout (bytes, all 256-aligned; ~192 MB total) ----
  char* ws = (char*)d_ws;
  size_t o = 0;
  bf16*  f_bf   = (bf16*)(ws + o);  o += (size_t)MROWS * CC * 2;   //  66.8 MB
  bf16*  q_bf   = (bf16*)(ws + o);  o += (size_t)BN * CC * 2;
  bf16*  Wvt    = (bf16*)(ws + o);  o += 256 * 256 * 2;
  bf16*  Woat   = (bf16*)(ws + o);  o += 384 * 256 * 2;
  bf16*  Woutt  = (bf16*)(ws + o);  o += 256 * 256 * 2;
  bf16*  value  = (bf16*)(ws + o);  o += (size_t)MROWS * CC * 2;   //  66.8 MB
  float* oa     = (float*)(ws + o); o += (size_t)BN * 384 * 4;
  float* aw     = (float*)(ws + o); o += (size_t)BN * 128 * 4;
  float* loc    = (float*)(ws + o); o += (size_t)BN * 256 * 4;
  bf16*  samp   = (bf16*)(ws + o);  o += (size_t)BN * CC * 2;
  float* attn   = (float*)(ws + o); o += (size_t)BN * CC * 4;
  float* l1f    = (float*)(ws + o); o += (size_t)BB * 3072 * CC * 4; // 25.2 MB
  (void)ws_size; (void)in_sizes; (void)n_in; (void)out_size;

  prep<<<2048, 256, 0, stream>>>(q_t, Wv, Woff, Wattn, Wout, q_bf, Wvt, Woat, Woutt);
  pyr01<<<(BB * 3072 * CC) / 256, 256, 0, stream>>>(h_t, f_bf, l1f);
  pyr23<<<(BB * 192 * CC) / 256, 256, 0, stream>>>(l1f, f_bf);
  gemm_n256<bf16><<<MROWS / 32, 128, 0, stream>>>(f_bf, Wvt, bv, value);   // value
  gemm_n384<<<BN / 16, 192, 0, stream>>>(q_bf, Woat, boff, battn, oa);     // off|attn
  locaw<<<(BN * 8) / 256, 256, 0, stream>>>(oa, tc, aw, loc);
  sample<<<(BN * 8) / 8, 256, 0, stream>>>(value, loc, aw, samp);
  gemm_n256<float><<<BN / 32, 128, 0, stream>>>(samp, Woutt, bout, attn);  // out proj
  finalize<<<BN / 8, 256, 0, stream>>>(q_t, attn, lng, lnb, Wvis, bvis, Wunc, bunc, out);
}
